// LSTM_76622216560744
// MI455X (gfx1250) — compile-verified, matrix operand order checked by reading, not run on hardware
//
#include <hip/hip_runtime.h>
#include <hip/hip_bf16.h>

// ---------------------------------------------------------------------------
// 2-layer LSTM (H=50, input=1), T=1024, B=2048, final linear head.
// Strategy (MI455X / gfx1250, wave32):
//   * batch split into 128 independent 16-row tiles; one wave per tile.
//   * weights (3 x [200x50]) pre-packed to f16 WMMA B-fragment layout,
//     resident in LDS for the whole scan.
//   * each step: gates = WMMA f32<-f16 (16x16x32, K padded 50->64),
//     D staged in wave-private LDS, fp32 elementwise LSTM cell update,
//     new h written back (f16) into WMMA A staging layout.
//   * 78 v_wmma per wave per timestep; no barriers inside the t-loop.
// ---------------------------------------------------------------------------

typedef __attribute__((ext_vector_type(16))) _Float16 v16h;
typedef __attribute__((ext_vector_type(8)))  _Float16 v8h;
typedef __attribute__((ext_vector_type(8)))  float    v8f;
typedef __attribute__((ext_vector_type(4)))  float    v4f;

#define LSTM_H     50
#define LSTM_T     1024
#define LSTM_B     2048
#define NTILES     13           // ceil(200/16) n-tiles of the 4H gate dim
#define KPAD       64           // K=50 padded to 2 chunks of 32
#define FRAG_HALFS 512          // one B fragment: 32 lanes * 16 halfs
#define WMAT_HALFS (NTILES * 2 * FRAG_HALFS)   // 13312 halfs = 26624 B

// d_ws layout (bytes) == shared-memory prefix (copied verbatim to LDS)
#define WS_W      0                         // 3 packed weight mats (79872 B)
#define WS_WIH1   (3 * WMAT_HALFS * 2)      // 79872: wih1 f32[208]
#define WS_B1     (WS_WIH1 + 208 * 4)       // b_ih1+b_hh1 f32[208]
#define WS_B2     (WS_B1  + 208 * 4)        // b_ih2+b_hh2 f32[208]
#define WS_WLIN   (WS_B2  + 208 * 4)        // w_lin f32[64]
#define WS_TOTAL  (WS_WLIN + 64 * 4)        // 82624 B (16B aligned)

// per-wave private LDS region
#define PW_GSC    0                         // gate scratch: 13 frags * 256 f32
#define PW_H1     (PW_GSC + NTILES * 256 * 4)      // 13312: h1 staging [16][64] f16
#define PW_H2     (PW_H1 + 16 * KPAD * 2)          // h2 staging
#define PW_C1     (PW_H2 + 16 * KPAD * 2)          // c1 [16][64] f32
#define PW_C2     (PW_C1 + 16 * KPAD * 4)
#define PW_XR     (PW_C2 + 16 * KPAD * 4)          // x broadcast [16] f32
#define PW_BYTES  (PW_XR + 64)                     // 25664 B
#define WAVES_PER_BLOCK 2

// ---------------------------------------------------------------------------
// Pack kernel: fp32 weights -> f16 B-fragment layout + fused bias vectors.
// B[k][n] = W[n][k]; fragment (nt, kc): lane holds n = nt*16 + (lane&15),
// half j maps to k = kc*32 + (j<8 ? j : j+8) + (lane>=16 ? 8 : 0).
// ---------------------------------------------------------------------------
__global__ void lstm_pack_kernel(const float* __restrict__ Whh1,
                                 const float* __restrict__ Wih2,
                                 const float* __restrict__ Whh2,
                                 const float* __restrict__ wih1,
                                 const float* __restrict__ bi1,
                                 const float* __restrict__ bh1,
                                 const float* __restrict__ bi2,
                                 const float* __restrict__ bh2,
                                 const float* __restrict__ wlin,
                                 char* __restrict__ ws) {
    const int tid = blockIdx.x * blockDim.x + threadIdx.x;
    const int nth = gridDim.x * blockDim.x;

    _Float16* wp = (_Float16*)(ws + WS_W);
    for (int p = tid; p < 3 * WMAT_HALFS; p += nth) {
        const int mat    = p / WMAT_HALFS;
        const int q      = p - mat * WMAT_HALFS;
        const int frag   = q >> 9;          // /512
        const int within = q & 511;
        const int lane   = within >> 4;
        const int j      = within & 15;
        const int ntile  = frag >> 1;
        const int kc     = frag & 1;
        const int n = ntile * 16 + (lane & 15);
        const int k = kc * 32 + (j < 8 ? j : j + 8) + ((lane >= 16) ? 8 : 0);
        const float* W = (mat == 0) ? Whh1 : ((mat == 1) ? Wih2 : Whh2);
        float v = (n < 4 * LSTM_H && k < LSTM_H) ? W[n * LSTM_H + k] : 0.0f;
        wp[p] = (_Float16)v;
    }
    float* f;
    f = (float*)(ws + WS_WIH1);
    for (int p = tid; p < 208; p += nth) f[p] = (p < 200) ? wih1[p] : 0.0f;
    f = (float*)(ws + WS_B1);
    for (int p = tid; p < 208; p += nth) f[p] = (p < 200) ? (bi1[p] + bh1[p]) : 0.0f;
    f = (float*)(ws + WS_B2);
    for (int p = tid; p < 208; p += nth) f[p] = (p < 200) ? (bi2[p] + bh2[p]) : 0.0f;
    f = (float*)(ws + WS_WLIN);
    for (int p = tid; p < 64; p += nth) f[p] = (p < LSTM_H) ? wlin[p] : 0.0f;
}

// ---------------------------------------------------------------------------
// Fragment load helpers (all from LDS; two 16-byte ds loads each)
// ---------------------------------------------------------------------------
__device__ __forceinline__ v16h load_afrag(const _Float16* hs, int lane, int kc) {
    const int m   = lane & 15;
    const int hi8 = (lane >> 4) << 3;
    union { v16h v; v8h h[2]; } u;
    const _Float16* p = hs + m * KPAD + kc * 32 + hi8;
    u.h[0] = *(const v8h*)(p);
    u.h[1] = *(const v8h*)(p + 16);
    return u.v;
}

__device__ __forceinline__ v16h load_bfrag(const _Float16* wmat, int lane, int frag) {
    union { v16h v; v8h h[2]; } u;
    const _Float16* p = wmat + frag * FRAG_HALFS + lane * 16;
    u.h[0] = *(const v8h*)(p);
    u.h[1] = *(const v8h*)(p + 8);
    return u.v;
}

// D fragment scratch is stored [frag][lane][8 floats]; element (m, n):
__device__ __forceinline__ float gate_at(const float* gsc, int m, int n) {
    const int frag = n >> 4;
    const int ln   = (n & 15) + ((m & 8) ? 16 : 0);
    const int r    = m & 7;
    return gsc[frag * 256 + ln * 8 + r];
}

__device__ __forceinline__ float sigf(float v) {
    return 1.0f / (1.0f + __expf(-v));
}

#define WMMA_F16(A, B, C) \
    __builtin_amdgcn_wmma_f32_16x16x32_f16(false, (A), false, (B), (short)0, (C), false, false)

// ---------------------------------------------------------------------------
// Main scan kernel: one wave == one 16-row batch tile, full T loop.
// ---------------------------------------------------------------------------
__global__ void __launch_bounds__(WAVES_PER_BLOCK * 32)
lstm_wmma_scan_kernel(const float* __restrict__ x,
                      const float* __restrict__ blin_p,
                      const char*  __restrict__ ws,
                      float* __restrict__ out) {
    extern __shared__ char smem[];
    const int tid  = threadIdx.x;
    const int lane = tid & 31;
    const int wave = tid >> 5;
    const int rowbase = blockIdx.x * (WAVES_PER_BLOCK * 16) + wave * 16;

    // cooperative copy of packed weights/bias prefix: global -> LDS (once)
    {
        const uint4* src = (const uint4*)ws;
        uint4* dst = (uint4*)smem;
        for (int i = tid; i < WS_TOTAL / 16; i += blockDim.x) dst[i] = src[i];
    }
    __syncthreads();

    const _Float16* Whh1 = (const _Float16*)(smem + WS_W);
    const _Float16* Wih2 = Whh1 + WMAT_HALFS;
    const _Float16* Whh2 = Whh1 + 2 * WMAT_HALFS;
    const float* sWih1 = (const float*)(smem + WS_WIH1);
    const float* sB1   = (const float*)(smem + WS_B1);
    const float* sB2   = (const float*)(smem + WS_B2);
    const float* sWlin = (const float*)(smem + WS_WLIN);

    char* wbase = smem + WS_TOTAL + wave * PW_BYTES;
    float*    gsc = (float*)(wbase + PW_GSC);
    _Float16* h1s = (_Float16*)(wbase + PW_H1);
    _Float16* h2s = (_Float16*)(wbase + PW_H2);
    float*    c1  = (float*)(wbase + PW_C1);
    float*    c2  = (float*)(wbase + PW_C2);
    float*    xr  = (float*)(wbase + PW_XR);

    // zero state (covers K padding 50..63 too)
    for (int i = lane; i < 16 * KPAD; i += 32) {
        h1s[i] = (_Float16)0.0f;
        h2s[i] = (_Float16)0.0f;
        c1[i]  = 0.0f;
        c2[i]  = 0.0f;
    }

    for (int t = 0; t < LSTM_T; ++t) {
        if (lane < 16)
            xr[lane] = x[(size_t)(rowbase + lane) * LSTM_T + t];

        // ---------------- layer 1: gates = h1 @ Whh1^T ----------------
        {
            v16h a0 = load_afrag(h1s, lane, 0);
            v16h a1 = load_afrag(h1s, lane, 1);
#pragma unroll
            for (int nt = 0; nt < NTILES; ++nt) {
                v8f acc = {};
                v16h b0 = load_bfrag(Whh1, lane, nt * 2 + 0);
                acc = WMMA_F16(a0, b0, acc);
                v16h b1 = load_bfrag(Whh1, lane, nt * 2 + 1);
                acc = WMMA_F16(a1, b1, acc);
                union { v8f v; v4f q[2]; } du; du.v = acc;
                v4f* dp = (v4f*)(gsc + nt * 256 + lane * 8);
                dp[0] = du.q[0];
                dp[1] = du.q[1];
            }
        }
        // layer-1 elementwise cell update (800 items == 25 full-EXEC iters)
#pragma unroll 5
        for (int it = 0; it < 25; ++it) {
            const int idx = it * 32 + lane;
            const int m = idx / LSTM_H;
            const int j = idx - m * LSTM_H;
            const float xm = xr[m];
            const float gi = gate_at(gsc, m, j)       + xm * sWih1[j]       + sB1[j];
            const float gf = gate_at(gsc, m, 50 + j)  + xm * sWih1[50 + j]  + sB1[50 + j];
            const float gg = gate_at(gsc, m, 100 + j) + xm * sWih1[100 + j] + sB1[100 + j];
            const float go = gate_at(gsc, m, 150 + j) + xm * sWih1[150 + j] + sB1[150 + j];
            const float c  = sigf(gf) * c1[m * KPAD + j] + sigf(gi) * tanhf(gg);
            c1[m * KPAD + j]  = c;
            h1s[m * KPAD + j] = (_Float16)(sigf(go) * tanhf(c));
        }

        // -------- layer 2: gates = h1 @ Wih2^T + h2 @ Whh2^T --------
        {
            v16h a0 = load_afrag(h1s, lane, 0);
            v16h a1 = load_afrag(h1s, lane, 1);
            v16h a2 = load_afrag(h2s, lane, 0);
            v16h a3 = load_afrag(h2s, lane, 1);
#pragma unroll
            for (int nt = 0; nt < NTILES; ++nt) {
                v8f acc = {};
                v16h b0 = load_bfrag(Wih2, lane, nt * 2 + 0);
                acc = WMMA_F16(a0, b0, acc);
                v16h b1 = load_bfrag(Wih2, lane, nt * 2 + 1);
                acc = WMMA_F16(a1, b1, acc);
                v16h b2 = load_bfrag(Whh2, lane, nt * 2 + 0);
                acc = WMMA_F16(a2, b2, acc);
                v16h b3 = load_bfrag(Whh2, lane, nt * 2 + 1);
                acc = WMMA_F16(a3, b3, acc);
                union { v8f v; v4f q[2]; } du; du.v = acc;
                v4f* dp = (v4f*)(gsc + nt * 256 + lane * 8);
                dp[0] = du.q[0];
                dp[1] = du.q[1];
            }
        }
        // layer-2 elementwise cell update
#pragma unroll 5
        for (int it = 0; it < 25; ++it) {
            const int idx = it * 32 + lane;
            const int m = idx / LSTM_H;
            const int j = idx - m * LSTM_H;
            const float gi = gate_at(gsc, m, j)       + sB2[j];
            const float gf = gate_at(gsc, m, 50 + j)  + sB2[50 + j];
            const float gg = gate_at(gsc, m, 100 + j) + sB2[100 + j];
            const float go = gate_at(gsc, m, 150 + j) + sB2[150 + j];
            const float c  = sigf(gf) * c2[m * KPAD + j] + sigf(gi) * tanhf(gg);
            c2[m * KPAD + j]  = c;
            h2s[m * KPAD + j] = (_Float16)(sigf(go) * tanhf(c));
        }
    }

    // final head: out[row] = h2 . w_lin + b_lin
    if (lane < 16) {
        float acc = blin_p[0];
#pragma unroll
        for (int j = 0; j < LSTM_H; ++j)
            acc += (float)h2s[lane * KPAD + j] * sWlin[j];
        out[rowbase + lane] = acc;
    }
}

// ---------------------------------------------------------------------------
extern "C" void kernel_launch(void* const* d_in, const int* in_sizes, int n_in,
                              void* d_out, int out_size, void* d_ws, size_t ws_size,
                              hipStream_t stream) {
    const float* x    = (const float*)d_in[0];
    const float* Wih1 = (const float*)d_in[1];
    const float* Whh1 = (const float*)d_in[2];
    const float* bih1 = (const float*)d_in[3];
    const float* bhh1 = (const float*)d_in[4];
    const float* Wih2 = (const float*)d_in[5];
    const float* Whh2 = (const float*)d_in[6];
    const float* bih2 = (const float*)d_in[7];
    const float* bhh2 = (const float*)d_in[8];
    const float* Wlin = (const float*)d_in[9];
    const float* blin = (const float*)d_in[10];

    lstm_pack_kernel<<<64, 256, 0, stream>>>(Whh1, Wih2, Whh2, Wih1, bih1, bhh1,
                                             bih2, bhh2, Wlin, (char*)d_ws);

    const size_t smem = WS_TOTAL + WAVES_PER_BLOCK * PW_BYTES;  // ~131 KB
    (void)hipFuncSetAttribute((const void*)lstm_wmma_scan_kernel,
                              hipFuncAttributeMaxDynamicSharedMemorySize,
                              (int)smem);
    const int blocks = LSTM_B / (WAVES_PER_BLOCK * 16);  // 64
    lstm_wmma_scan_kernel<<<blocks, WAVES_PER_BLOCK * 32, smem, stream>>>(
        x, blin, (const char*)d_ws, (float*)d_out);
}
